// MultKAN_63462436766154
// MI455X (gfx1250) — matrix operand order repaired and checked by reading, not compiled, not run on hardware
//
#include <hip/hip_runtime.h>
#include <hip/hip_bf16.h>

typedef __bf16 bf16_t;
typedef __attribute__((ext_vector_type(16))) __bf16 v16bf;
typedef __attribute__((ext_vector_type(8)))  float  v8f;

#define BATCH 4096

// ---------------------------------------------------------------------------
// Fragment layouts (CDNA5 ISA 7.12.2, wave32):
//  A (16x32 MxK, bf16):  lane = (m&15) | (((k>>3)&1)<<4),  elem e = ((k>>4)<<3)|(k&7)
//  B (32x16 KxN, bf16):  lane = (n&15) | (((k>>3)&1)<<4),  elem e = ((k>>4)<<3)|(k&7)
//  D (16x16 f32):        lane = (n&15) | (hi<<4), vgpr r -> m = r + 8*hi
// Tiles stored as [tile][lane(32)][elem(16)] so the GEMM does contiguous
// 32-byte per-lane loads (global_load_b128 x2) straight into WMMA operands.
// ---------------------------------------------------------------------------

// Pack W[(i*9+s), o] = (s==0 ? sb[i,o] : coef[i,o,s-1]*ss[i,o]) into B-fragments.
__global__ void pack_w_kernel(const float* __restrict__ sb,
                              const float* __restrict__ coef,
                              const float* __restrict__ ss,
                              __hip_bfloat16* __restrict__ W,
                              int din, int dout, int Kt) {
    int idx = blockIdx.x * blockDim.x + threadIdx.x;
    int i = idx / dout;
    int o = idx - i * dout;
    if (i >= din) return;

    float ssv = ss[(size_t)i * dout + o];
    float vals[9];
    vals[0] = sb[(size_t)i * dout + o];
    const float* cp = coef + ((size_t)i * dout + o) * 8;
#pragma unroll
    for (int g = 0; g < 8; ++g) vals[1 + g] = cp[g] * ssv;

    int nt = o >> 4;
    int lm = o & 15;
#pragma unroll
    for (int s = 0; s < 9; ++s) {
        int c  = i * 9 + s;
        int kt = c >> 5;
        int kk = c & 31;
        int hi = (kk >> 3) & 1;
        int e  = ((kk >> 4) << 3) | (kk & 7);
        int lane = lm | (hi << 4);
        W[((((size_t)nt * Kt + kt) << 5) + lane) * 16 + e] = __float2bfloat16(vals[s]);
    }
}

// SiLU + cubic B-spline basis (G=5,K=3 -> 12 grid pts, 8 basis fns), packed
// into A-fragments as bf16.
__global__ void act_pack_kernel(const float* __restrict__ x,
                                const float* __restrict__ grid,
                                __hip_bfloat16* __restrict__ A,
                                int din, int Kt) {
    int idx = blockIdx.x * blockDim.x + threadIdx.x;
    int b = idx / din;
    int i = idx - b * din;

    float xv = x[(size_t)b * din + i];

    float g[12];
#pragma unroll
    for (int j = 0; j < 12; ++j) g[j] = grid[(size_t)i * 12 + j];

    float v[11];
#pragma unroll
    for (int j = 0; j < 11; ++j)
        v[j] = (xv >= g[j] && xv < g[j + 1]) ? 1.0f : 0.0f;
#pragma unroll
    for (int p = 1; p <= 3; ++p) {
#pragma unroll
        for (int j = 0; j < 8; ++j) {
            v[j] = (xv - g[j]) / (g[j + p] - g[j]) * v[j]
                 + (g[j + p + 1] - xv) / (g[j + p + 1] - g[j + 1]) * v[j + 1];
        }
        if (p == 1) { v[8] = (xv - g[8])/(g[9]-g[8])*v[8] + (g[10]-xv)/(g[10]-g[9])*v[9];
                      v[9] = (xv - g[9])/(g[10]-g[9])*v[9] + (g[11]-xv)/(g[11]-g[10])*v[10]; }
        if (p == 2) { v[8] = (xv - g[8])/(g[10]-g[8])*v[8] + (g[11]-xv)/(g[11]-g[9])*v[9]; }
    }

    float vals[9];
    vals[0] = xv / (1.0f + __expf(-xv));   // SiLU
#pragma unroll
    for (int s = 0; s < 8; ++s) vals[1 + s] = v[s];

    int mt = b >> 4;
    int lm = b & 15;
#pragma unroll
    for (int s = 0; s < 9; ++s) {
        int c  = i * 9 + s;
        int kt = c >> 5;
        int kk = c & 31;
        int lane = lm | (((kk >> 3) & 1) << 4);
        int e    = ((kk >> 4) << 3) | (kk & 7);
        A[((((size_t)mt * Kt + kt) << 5) + lane) * 16 + e] = __float2bfloat16(vals[s]);
    }
}

__device__ __forceinline__ void wmma8(v8f acc[2][4],
                                      v16bf a0, v16bf a1,
                                      v16bf b0, v16bf b1, v16bf b2, v16bf b3) {
    acc[0][0] = __builtin_amdgcn_wmma_f32_16x16x32_bf16(false, a0, false, b0, (short)0, acc[0][0], false, false);
    acc[0][1] = __builtin_amdgcn_wmma_f32_16x16x32_bf16(false, a0, false, b1, (short)0, acc[0][1], false, false);
    acc[0][2] = __builtin_amdgcn_wmma_f32_16x16x32_bf16(false, a0, false, b2, (short)0, acc[0][2], false, false);
    acc[0][3] = __builtin_amdgcn_wmma_f32_16x16x32_bf16(false, a0, false, b3, (short)0, acc[0][3], false, false);
    acc[1][0] = __builtin_amdgcn_wmma_f32_16x16x32_bf16(false, a1, false, b0, (short)0, acc[1][0], false, false);
    acc[1][1] = __builtin_amdgcn_wmma_f32_16x16x32_bf16(false, a1, false, b1, (short)0, acc[1][1], false, false);
    acc[1][2] = __builtin_amdgcn_wmma_f32_16x16x32_bf16(false, a1, false, b2, (short)0, acc[1][2], false, false);
    acc[1][3] = __builtin_amdgcn_wmma_f32_16x16x32_bf16(false, a1, false, b3, (short)0, acc[1][3], false, false);
}

// bf16 WMMA GEMM: Y[M,N] = A[M,K9] * W[K9,N].  One wave per 32x64 output
// block (2 M-tiles x 4 N-tiles), 8 v_wmma_f32_16x16x32_bf16 per k-step.
// Manually 2-stage software-pipelined over pairs of k-tiles (Kt is even):
// two fixed fragment register sets alternate load/compute roles, so there
// are no rotation moves and no compiler re-unrolling (#pragma unroll 1).
__global__ void __launch_bounds__(256)
gemm_bf16_kernel(const bf16_t* __restrict__ A,
                 const bf16_t* __restrict__ W,
                 float* __restrict__ Y,
                 int Kt, int N, int Ng) {
    int wave = threadIdx.x >> 5;
    int lane = threadIdx.x & 31;
    int group = blockIdx.x * 8 + wave;
    int mg = group / Ng;
    int ng = group - mg * Ng;
    int mt0 = mg << 1;   // 2 M-tiles
    int nt0 = ng << 2;   // 4 N-tiles

    const bf16_t* pA0 = A + (((size_t)(mt0 + 0) * Kt) << 9) + (lane << 4);
    const bf16_t* pA1 = A + (((size_t)(mt0 + 1) * Kt) << 9) + (lane << 4);
    const bf16_t* pB0 = W + (((size_t)(nt0 + 0) * Kt) << 9) + (lane << 4);
    const bf16_t* pB1 = W + (((size_t)(nt0 + 1) * Kt) << 9) + (lane << 4);
    const bf16_t* pB2 = W + (((size_t)(nt0 + 2) * Kt) << 9) + (lane << 4);
    const bf16_t* pB3 = W + (((size_t)(nt0 + 3) * Kt) << 9) + (lane << 4);

    v8f acc[2][4];
#pragma unroll
    for (int m = 0; m < 2; ++m)
#pragma unroll
        for (int n = 0; n < 4; ++n)
#pragma unroll
            for (int r = 0; r < 8; ++r) acc[m][n][r] = 0.0f;

    // stage-0 set: even k-tile
    v16bf a0 = *(const v16bf*)pA0;
    v16bf a1 = *(const v16bf*)pA1;
    v16bf b0 = *(const v16bf*)pB0;
    v16bf b1 = *(const v16bf*)pB1;
    v16bf b2 = *(const v16bf*)pB2;
    v16bf b3 = *(const v16bf*)pB3;

    int pairs = (Kt >> 1) - 1;           // Kt is even (144 / 162)
#pragma unroll 1
    for (int t = 0; t < pairs; ++t) {
        // stage-1 set: odd k-tile (compile-time +512 elem = +1024B immediate)
        v16bf c0 = *(const v16bf*)(pA0 + 512);
        v16bf c1 = *(const v16bf*)(pA1 + 512);
        v16bf d0 = *(const v16bf*)(pB0 + 512);
        v16bf d1 = *(const v16bf*)(pB1 + 512);
        v16bf d2 = *(const v16bf*)(pB2 + 512);
        v16bf d3 = *(const v16bf*)(pB3 + 512);
        __builtin_prefetch(pA0 + 2048, 0, 1);   // global_prefetch_b8, 4 tiles ahead
        __builtin_prefetch(pA1 + 2048, 0, 1);

        wmma8(acc, a0, a1, b0, b1, b2, b3);     // even tile 2t

        pA0 += 1024; pA1 += 1024;
        pB0 += 1024; pB1 += 1024; pB2 += 1024; pB3 += 1024;

        // stage-0 set: next even k-tile 2t+2
        a0 = *(const v16bf*)pA0;
        a1 = *(const v16bf*)pA1;
        b0 = *(const v16bf*)pB0;
        b1 = *(const v16bf*)pB1;
        b2 = *(const v16bf*)pB2;
        b3 = *(const v16bf*)pB3;

        wmma8(acc, c0, c1, d0, d1, d2, d3);     // odd tile 2t+1
    }

    // tail pair: tiles Kt-2 (in stage-0 set) and Kt-1
    {
        v16bf c0 = *(const v16bf*)(pA0 + 512);
        v16bf c1 = *(const v16bf*)(pA1 + 512);
        v16bf d0 = *(const v16bf*)(pB0 + 512);
        v16bf d1 = *(const v16bf*)(pB1 + 512);
        v16bf d2 = *(const v16bf*)(pB2 + 512);
        v16bf d3 = *(const v16bf*)(pB3 + 512);

        wmma8(acc, a0, a1, b0, b1, b2, b3);
        wmma8(acc, c0, c1, d0, d1, d2, d3);
    }

    int hi = lane >> 4;
    int lm = lane & 15;
#pragma unroll
    for (int m = 0; m < 2; ++m) {
#pragma unroll
        for (int n = 0; n < 4; ++n) {
            float* yp = Y + ((size_t)((mt0 + m) * 16 + 8 * hi)) * N + (nt0 + n) * 16 + lm;
#pragma unroll
            for (int r = 0; r < 8; ++r) yp[(size_t)r * N] = acc[m][n][r];
        }
    }
}

// subnode affine -> pair product for mult nodes -> node affine
__global__ void epilogue_kernel(const float* __restrict__ Y,
                                const float* __restrict__ snb,
                                const float* __restrict__ sns,
                                const float* __restrict__ nb,
                                const float* __restrict__ ns,
                                float* __restrict__ Xout,
                                int dout, int n_add, int dnode) {
    int idx = blockIdx.x * blockDim.x + threadIdx.x;
    int b = idx / dnode;
    int o = idx - b * dnode;
    const float* yr = Y + (size_t)b * dout;
    float u;
    if (o < n_add) {
        u = sns[o] * yr[o] + snb[o];
    } else {
        int o1 = n_add + 2 * (o - n_add);
        float u1 = sns[o1]     * yr[o1]     + snb[o1];
        float u2 = sns[o1 + 1] * yr[o1 + 1] + snb[o1 + 1];
        u = u1 * u2;
    }
    Xout[(size_t)b * dnode + o] = ns[o] * u + nb[o];
}

extern "C" void kernel_launch(void* const* d_in, const int* in_sizes, int n_in,
                              void* d_out, int out_size, void* d_ws, size_t ws_size,
                              hipStream_t stream) {
    (void)in_sizes; (void)n_in; (void)out_size; (void)ws_size;

    const int din_a[3]   = {512, 576, 576};
    const int dout_a[3]  = {640, 640, 128};
    const int dnode_a[3] = {576, 576, 128};
    const int nadd_a[3]  = {512, 512, 128};

    char* ws = (char*)d_ws;
    __hip_bfloat16* Apack = (__hip_bfloat16*)(ws);                       // <= 42.5 MB
    __hip_bfloat16* Wpack = (__hip_bfloat16*)(ws + 44040192);            // <= 6.7 MB
    float*          Ybuf  = (float*)(ws + 44040192 + 7340032);           // <= 10.5 MB
    float*          Xbuf  = (float*)(ws + 44040192 + 7340032 + 11534336);// <= 9.4 MB

    const float* xin = (const float*)d_in[0];

    for (int l = 0; l < 3; ++l) {
        const float* grid = (const float*)d_in[1 + 8 * l + 0];
        const float* coef = (const float*)d_in[1 + 8 * l + 1];
        const float* sb   = (const float*)d_in[1 + 8 * l + 2];
        const float* ss   = (const float*)d_in[1 + 8 * l + 3];
        const float* snb  = (const float*)d_in[1 + 8 * l + 4];
        const float* sns  = (const float*)d_in[1 + 8 * l + 5];
        const float* nb   = (const float*)d_in[1 + 8 * l + 6];
        const float* ns   = (const float*)d_in[1 + 8 * l + 7];

        int din = din_a[l], dout = dout_a[l], dnode = dnode_a[l], n_add = nadd_a[l];
        int K9 = din * 9;
        int Kt = K9 / 32;               // 144 / 162 / 162  (exact, even)

        pack_w_kernel<<<(din * dout) / 256, 256, 0, stream>>>(sb, coef, ss, Wpack, din, dout, Kt);
        act_pack_kernel<<<(BATCH * din) / 256, 256, 0, stream>>>(xin, grid, Apack, din, Kt);

        int Ng = dout / 64;                       // N-groups of 4 tiles
        int groups = (BATCH / 32) * Ng;           // M-groups of 2 tiles
        gemm_bf16_kernel<<<groups / 8, 256, 0, stream>>>((const bf16_t*)Apack, (const bf16_t*)Wpack,
                                                         Ybuf, Kt, dout, Ng);

        float* xo = (l == 2) ? (float*)d_out : Xbuf;
        epilogue_kernel<<<(BATCH * dnode) / 256, 256, 0, stream>>>(Ybuf, snb, sns, nb, ns, xo,
                                                                   dout, n_add, dnode);
        xin = Xbuf;
    }
}
